// HiddenStateDiscretizer_48919677501656
// MI455X (gfx1250) — matrix operand order repaired and checked by reading, not compiled, not run on hardware
//
#include <hip/hip_runtime.h>
#include <hip/hip_bf16.h>

// ---------------------------------------------------------------------------
// Types
// ---------------------------------------------------------------------------
typedef __bf16 bf16;
typedef __attribute__((ext_vector_type(16))) __bf16 bf16x16;
typedef __attribute__((ext_vector_type(8)))  __bf16 bf16x8;
typedef __attribute__((ext_vector_type(4)))  __bf16 bf16x4;
typedef __attribute__((ext_vector_type(8)))  float  f32x8;

#define BATCH     16384
#define HIDDEN    2048
#define FC1       1024
#define FC2       512
#define CODE      256
#define KBOOK     512
#define NEG_SLOPE 0.01f
#define BN_EPS    1e-5f
#define BN_CHUNKS 64

// ---------------------------------------------------------------------------
// WMMA fragment loads (CDNA5 ISA 7.12.2 layouts, wave32)
// A 16x32 bf16: lanes 0-15 -> M=lane, elems[0..7]=K0..7, elems[8..15]=K16..23
//               lanes16-31 -> M=lane-16, K offsets +8
// B 32x16 bf16: lanes 0-15 -> N=lane, elems[0..7]=K0..7, elems[8..15]=K16..23; lanes 16-31 K offsets +8
// Both are contiguous b128 loads because B is stored transposed [N,K].
// ---------------------------------------------------------------------------
__device__ __forceinline__ bf16x16 frag_a(const bf16* __restrict__ p, int hi) {
  bf16x8 lo = *(const bf16x8*)(p + hi * 8);
  bf16x8 hh = *(const bf16x8*)(p + 16 + hi * 8);
  bf16x16 r;
#pragma unroll
  for (int i = 0; i < 8; ++i) { r[i] = lo[i]; r[i + 8] = hh[i]; }
  return r;
}

__device__ __forceinline__ bf16x16 frag_b(const bf16* __restrict__ p, int hi) {
  bf16x8 lo = *(const bf16x8*)(p + hi * 8);
  bf16x8 hh = *(const bf16x8*)(p + 16 + hi * 8);
  bf16x16 r;
#pragma unroll
  for (int i = 0; i < 8; ++i) { r[i] = lo[i]; r[i + 8] = hh[i]; }
  return r;
}

// ---------------------------------------------------------------------------
// GEMM: C[M,N] = A[M,K](bf16) * Bt[N,K](bf16)^T (+bias), f32 accum.
// Block = 8 waves (2 x 4); wave tile = 32x64 (2x4 WMMA tiles, 8 accums)
// => block tile 64x256. grid = (M/64, N/256). K % 32 == 0.
// Per 32-K step: 12 b128 loads feed 8 v_wmma_f32_16x16x32_bf16.
// ---------------------------------------------------------------------------
__global__ __launch_bounds__(256)
void gemm_bf16_wmma(const bf16* __restrict__ A, const bf16* __restrict__ Bt,
                    const float* __restrict__ bias,
                    float* __restrict__ outF, bf16* __restrict__ outB,
                    int M, int N, int K)
{
  const int lane = threadIdx.x & 31;
  const int wid  = threadIdx.x >> 5;
  const int wm   = wid >> 2;          // 0..1
  const int wn   = wid & 3;           // 0..3
  const int l15  = lane & 15;
  const int hi   = lane >> 4;

  const long bm = (long)blockIdx.x * 64 + wm * 32;
  const long bn = (long)blockIdx.y * 256 + wn * 64;

  f32x8 acc[2][4] = {};

  const bf16* a0 = A + (bm + l15) * (long)K;
  const bf16* a1 = a0 + 16L * K;
  const bf16* bp[4];
#pragma unroll
  for (int j = 0; j < 4; ++j) bp[j] = Bt + (bn + j * 16 + l15) * (long)K;

  for (int kb = 0; kb < K; kb += 32) {
    bf16x16 fa0 = frag_a(a0 + kb, hi);
    bf16x16 fa1 = frag_a(a1 + kb, hi);
    bf16x16 fb[4];
#pragma unroll
    for (int j = 0; j < 4; ++j) fb[j] = frag_b(bp[j] + kb, hi);
#pragma unroll
    for (int j = 0; j < 4; ++j) {
      acc[0][j] = __builtin_amdgcn_wmma_f32_16x16x32_bf16(
          false, fa0, false, fb[j], (short)0, acc[0][j], false, false);
      acc[1][j] = __builtin_amdgcn_wmma_f32_16x16x32_bf16(
          false, fa1, false, fb[j], (short)0, acc[1][j], false, false);
    }
  }

  // C/D layout: elem i -> row = base + i + hi*8, col = base + (lane&15)
#pragma unroll
  for (int im = 0; im < 2; ++im) {
#pragma unroll
    for (int jn = 0; jn < 4; ++jn) {
      const long col = bn + jn * 16 + l15;
      const float bv = bias ? bias[col] : 0.0f;
#pragma unroll
      for (int i = 0; i < 8; ++i) {
        const long row = bm + im * 16 + hi * 8 + i;
        float v = acc[im][jn][i] + bv;
        if (outF) outF[row * (long)N + col] = v;
        if (outB) outB[row * (long)N + col] = (bf16)v;
      }
    }
  }
}

// ---------------------------------------------------------------------------
// Weight convert + transpose: W f32 [K,N] -> Wt bf16 [N,K]
// ---------------------------------------------------------------------------
__global__ void wt_transpose(const float* __restrict__ W, bf16* __restrict__ Wt,
                             int K, int N)
{
  long i = (long)blockIdx.x * blockDim.x + threadIdx.x;
  long total = (long)K * N;
  if (i >= total) return;
  long n = i / K, k = i % K;
  Wt[i] = (bf16)W[k * (long)N + n];
}

// f32 -> bf16 elementwise (vec4)
__global__ void f32_to_bf16_vec(const float* __restrict__ in, bf16* __restrict__ out,
                                long total4)
{
  long i = (long)blockIdx.x * blockDim.x + threadIdx.x;
  if (i >= total4) return;
  float4 v = ((const float4*)in)[i];
  bf16x4 r = {(bf16)v.x, (bf16)v.y, (bf16)v.z, (bf16)v.w};
  ((bf16x4*)out)[i] = r;
}

// ---------------------------------------------------------------------------
// BatchNorm stats (training mode, biased var): deterministic two-pass.
// ---------------------------------------------------------------------------
__global__ void bn_partial(const float* __restrict__ Y, float* __restrict__ ps,
                           float* __restrict__ pq, int M, int N)
{
  int c = blockIdx.x * blockDim.x + threadIdx.x;
  if (c >= N) return;
  const int rows = M / BN_CHUNKS;
  const long r0 = (long)blockIdx.y * rows;
  float s = 0.f, q = 0.f;
  for (int r = 0; r < rows; ++r) {
    float v = Y[(r0 + r) * (long)N + c];
    s += v; q += v * v;
  }
  ps[(long)blockIdx.y * N + c] = s;
  pq[(long)blockIdx.y * N + c] = q;
}

__global__ void bn_finalize(const float* __restrict__ ps, const float* __restrict__ pq,
                            const float* __restrict__ g, const float* __restrict__ b,
                            float* __restrict__ scale, float* __restrict__ shift,
                            int M, int N)
{
  int c = blockIdx.x * blockDim.x + threadIdx.x;
  if (c >= N) return;
  float s = 0.f, q = 0.f;
  for (int ch = 0; ch < BN_CHUNKS; ++ch) {
    s += ps[(long)ch * N + c];
    q += pq[(long)ch * N + c];
  }
  float mu  = s / (float)M;
  float var = q / (float)M - mu * mu;
  float sc  = g[c] * rsqrtf(var + BN_EPS);
  scale[c] = sc;
  shift[c] = b[c] - mu * sc;
}

// normalize + LeakyReLU + store bf16 for next GEMM
__global__ void bn_lrelu_bf16(const float* __restrict__ Y,
                              const float* __restrict__ scale,
                              const float* __restrict__ shift,
                              bf16* __restrict__ out, long total4, int N4)
{
  long i = (long)blockIdx.x * blockDim.x + threadIdx.x;
  if (i >= total4) return;
  float4 v = ((const float4*)Y)[i];
  int c = (int)(i % N4) * 4;
  float x0 = v.x * scale[c + 0] + shift[c + 0];
  float x1 = v.y * scale[c + 1] + shift[c + 1];
  float x2 = v.z * scale[c + 2] + shift[c + 2];
  float x3 = v.w * scale[c + 3] + shift[c + 3];
  x0 = x0 >= 0.f ? x0 : NEG_SLOPE * x0;
  x1 = x1 >= 0.f ? x1 : NEG_SLOPE * x1;
  x2 = x2 >= 0.f ? x2 : NEG_SLOPE * x2;
  x3 = x3 >= 0.f ? x3 : NEG_SLOPE * x3;
  bf16x4 r = {(bf16)x0, (bf16)x1, (bf16)x2, (bf16)x3};
  ((bf16x4*)out)[i] = r;
}

// ---------------------------------------------------------------------------
// Codebook column squared norms: cb f32 [CODE, KBOOK]
// ---------------------------------------------------------------------------
__global__ void codebook_norms(const float* __restrict__ cb, float* __restrict__ norms)
{
  int k = blockIdx.x * blockDim.x + threadIdx.x;
  if (k >= KBOOK) return;
  float s = 0.f;
  for (int d = 0; d < CODE; ++d) {
    float v = cb[(long)d * KBOOK + k];
    s += v * v;
  }
  norms[k] = s;
}

// ---------------------------------------------------------------------------
// Per-row argmin over k of (||c_k||^2 - 2 * S[b,k]); gather z_q from f32
// codebook; write f32 z_q + float(idx) to d_out and bf16 z_q for decoder.
// One 256-thread block per batch row. First-index tie-break (matches argmin).
// ---------------------------------------------------------------------------
__global__ __launch_bounds__(256)
void argmin_gather(const float* __restrict__ S, const float* __restrict__ cbn,
                   const float* __restrict__ cb, float* __restrict__ zq_out,
                   float* __restrict__ idx_out, bf16* __restrict__ zq_b)
{
  __shared__ float sv[256];
  __shared__ int   si[256];
  const int b = blockIdx.x;
  const int t = threadIdx.x;

  float best = 3.402823e38f;
  int   bi   = 0;
  for (int k = t; k < KBOOK; k += 256) {
    float d = cbn[k] - 2.0f * S[(long)b * KBOOK + k];
    if (d < best) { best = d; bi = k; }
  }
  sv[t] = best; si[t] = bi;
  __syncthreads();
#pragma unroll
  for (int s = 128; s > 0; s >>= 1) {
    if (t < s) {
      if (sv[t + s] < sv[t] || (sv[t + s] == sv[t] && si[t + s] < si[t])) {
        sv[t] = sv[t + s]; si[t] = si[t + s];
      }
    }
    __syncthreads();
  }
  const int idx = si[0];
  if (t == 0) idx_out[b] = (float)idx;
  // gather: z_q[b, t] = codebook[t, idx] (CODE == blockDim)
  float v = cb[(long)t * KBOOK + idx];
  zq_out[(long)b * CODE + t] = v;
  zq_b [(long)b * CODE + t] = (bf16)v;
}

// ---------------------------------------------------------------------------
// Host-side orchestration
// ---------------------------------------------------------------------------
static inline void launch_gemm(const bf16* A, const bf16* Bt, const float* bias,
                               float* outF, bf16* outB, int M, int N, int K,
                               hipStream_t s)
{
  dim3 grid(M / 64, N / 256);
  gemm_bf16_wmma<<<grid, 256, 0, s>>>(A, Bt, bias, outF, outB, M, N, K);
}

static inline void launch_bn(const float* Y, const float* g, const float* b,
                             float* ps, float* pq, float* scale, float* shift,
                             int M, int N, hipStream_t s)
{
  dim3 pg((N + 255) / 256, BN_CHUNKS);
  bn_partial<<<pg, 256, 0, s>>>(Y, ps, pq, M, N);
  bn_finalize<<<(N + 255) / 256, 256, 0, s>>>(ps, pq, g, b, scale, shift, M, N);
}

extern "C" void kernel_launch(void* const* d_in, const int* in_sizes, int n_in,
                              void* d_out, int out_size, void* d_ws, size_t ws_size,
                              hipStream_t stream)
{
  (void)in_sizes; (void)n_in; (void)out_size; (void)ws_size;

  const float* h        = (const float*)d_in[0];
  const float* enc_w1   = (const float*)d_in[1];
  const float* enc_g1   = (const float*)d_in[2];
  const float* enc_b1   = (const float*)d_in[3];
  const float* enc_w2   = (const float*)d_in[4];
  const float* enc_g2   = (const float*)d_in[5];
  const float* enc_b2   = (const float*)d_in[6];
  const float* enc_w3   = (const float*)d_in[7];
  const float* enc_b3   = (const float*)d_in[8];
  const float* codebook = (const float*)d_in[9];
  const float* dec_w1   = (const float*)d_in[10];
  const float* dec_g1   = (const float*)d_in[11];
  const float* dec_b1   = (const float*)d_in[12];
  const float* dec_w2   = (const float*)d_in[13];
  const float* dec_g2   = (const float*)d_in[14];
  const float* dec_b2   = (const float*)d_in[15];
  const float* dec_w3   = (const float*)d_in[16];
  const float* dec_b3   = (const float*)d_in[17];

  // d_out layout (f32): z_e | z_q | h_recon | indices(as float)
  float* out     = (float*)d_out;
  float* ze_out  = out;                                   // 16384*256
  float* zq_out  = out + (long)BATCH * CODE;              // 16384*256
  float* hr_out  = out + 2L * BATCH * CODE;               // 16384*2048
  float* idx_out = out + 2L * BATCH * CODE + (long)BATCH * HIDDEN;

  // workspace layout
  char* W = (char*)d_ws;
  bf16* Abuf = (bf16*)W;                         // 64 MiB activation arena (bf16)
  bf16* Aze  = Abuf + (16L << 20);               // +32 MiB (elements): z_e bf16
  bf16* Azq  = Abuf + (24L << 20);               // +48 MiB: z_q bf16
  float* Ybuf = (float*)(W + (64L << 20));       // 64 MiB f32 GEMM output arena
  bf16* Wts  = (bf16*)(W + (128L << 20));        // transposed bf16 weights
  bf16* w1t  = Wts;                              // [1024,2048]
  bf16* w2t  = w1t + 2097152;                    // [512,1024]
  bf16* w3t  = w2t + 524288;                     // [256,512]
  bf16* cbt  = w3t + 131072;                     // [512,256]
  bf16* wd1t = cbt + 131072;                     // [512,256]
  bf16* wd2t = wd1t + 131072;                    // [1024,512]
  bf16* wd3t = wd2t + 524288;                    // [2048,1024]
  float* scl = (float*)(W + (128L << 20) + 11272192);
  float* shf = scl + 1024;
  float* cbn = shf + 1024;
  float* ps  = cbn + 512;                        // 64*1024 partial sums
  float* pq  = ps + 65536;

  // --- one-time per launch: weights -> transposed bf16; h -> bf16 ----------
  wt_transpose<<<(HIDDEN * FC1) / 256, 256, 0, stream>>>(enc_w1, w1t, HIDDEN, FC1);
  wt_transpose<<<(FC1 * FC2) / 256, 256, 0, stream>>>(enc_w2, w2t, FC1, FC2);
  wt_transpose<<<(FC2 * CODE) / 256, 256, 0, stream>>>(enc_w3, w3t, FC2, CODE);
  wt_transpose<<<(CODE * KBOOK) / 256, 256, 0, stream>>>(codebook, cbt, CODE, KBOOK);
  wt_transpose<<<(CODE * FC2) / 256, 256, 0, stream>>>(dec_w1, wd1t, CODE, FC2);
  wt_transpose<<<(FC2 * FC1) / 256, 256, 0, stream>>>(dec_w2, wd2t, FC2, FC1);
  wt_transpose<<<(FC1 * HIDDEN) / 256, 256, 0, stream>>>(dec_w3, wd3t, FC1, HIDDEN);
  f32_to_bf16_vec<<<(long)BATCH * HIDDEN / 4 / 256, 256, 0, stream>>>(
      h, Abuf, (long)BATCH * HIDDEN / 4);

  // --- encoder -------------------------------------------------------------
  launch_gemm(Abuf, w1t, nullptr, Ybuf, nullptr, BATCH, FC1, HIDDEN, stream);
  launch_bn(Ybuf, enc_g1, enc_b1, ps, pq, scl, shf, BATCH, FC1, stream);
  bn_lrelu_bf16<<<(long)BATCH * FC1 / 4 / 256, 256, 0, stream>>>(
      Ybuf, scl, shf, Abuf, (long)BATCH * FC1 / 4, FC1 / 4);

  launch_gemm(Abuf, w2t, nullptr, Ybuf, nullptr, BATCH, FC2, FC1, stream);
  launch_bn(Ybuf, enc_g2, enc_b2, ps, pq, scl, shf, BATCH, FC2, stream);
  bn_lrelu_bf16<<<(long)BATCH * FC2 / 4 / 256, 256, 0, stream>>>(
      Ybuf, scl, shf, Abuf, (long)BATCH * FC2 / 4, FC2 / 4);

  // z_e = A2 @ w3 + b3  -> f32 to d_out, bf16 copy for quantization
  launch_gemm(Abuf, w3t, enc_b3, ze_out, Aze, BATCH, CODE, FC2, stream);

  // --- quantization --------------------------------------------------------
  codebook_norms<<<KBOOK / 256, 256, 0, stream>>>(codebook, cbn);
  launch_gemm(Aze, cbt, nullptr, Ybuf /*S[B,512]*/, nullptr, BATCH, KBOOK, CODE, stream);
  argmin_gather<<<BATCH, 256, 0, stream>>>(Ybuf, cbn, codebook, zq_out, idx_out, Azq);

  // --- decoder -------------------------------------------------------------
  launch_gemm(Azq, wd1t, nullptr, Ybuf, nullptr, BATCH, FC2, CODE, stream);
  launch_bn(Ybuf, dec_g1, dec_b1, ps, pq, scl, shf, BATCH, FC2, stream);
  bn_lrelu_bf16<<<(long)BATCH * FC2 / 4 / 256, 256, 0, stream>>>(
      Ybuf, scl, shf, Abuf, (long)BATCH * FC2 / 4, FC2 / 4);

  launch_gemm(Abuf, wd2t, nullptr, Ybuf, nullptr, BATCH, FC1, FC2, stream);
  launch_bn(Ybuf, dec_g2, dec_b2, ps, pq, scl, shf, BATCH, FC1, stream);
  bn_lrelu_bf16<<<(long)BATCH * FC1 / 4 / 256, 256, 0, stream>>>(
      Ybuf, scl, shf, Abuf, (long)BATCH * FC1 / 4, FC1 / 4);

  launch_gemm(Abuf, wd3t, dec_b3, hr_out, nullptr, BATCH, HIDDEN, FC1, stream);
}
